// HMSEncoder_7988639170754
// MI455X (gfx1250) — matrix operand order repaired and checked by reading, not compiled, not run on hardware
//
#include <hip/hip_runtime.h>

// ---------------------------------------------------------------------------
// Problem constants (from reference): V=30000 E=128 H=512 S=6 B=64 L=64
// ---------------------------------------------------------------------------
#define S_ 6
#define B_ 64
#define L_ 64
#define E_ 128
#define H_ 512
#define G_ 1536      // 3*H
#define V_ 30000

typedef __bf16 bf16;
typedef __attribute__((ext_vector_type(8)))  __bf16 v8bf;
typedef __attribute__((ext_vector_type(16))) __bf16 v16bf;
typedef __attribute__((ext_vector_type(8)))  float  v8f;

// ---------------------------------------------------------------------------
// WMMA operand loaders (CDNA5 v_wmma_f32_16x16x32_bf16 layouts, ISA 7.12.2)
// A (16x32, MxK): lanes 0-15 = rows, hold K {0..7, 16..23}; lanes 16-31 hold
// K {8..15, 24..31} for the same rows.
// B (32x16, KxN): lane n (0-15) holds K {0..7, 16..23} of column n; lane n+16 holds
// K {8..15, 24..31}. Our weights W are (N,K) row-major, so column n of B == row n of W.
// ---------------------------------------------------------------------------
__device__ __forceinline__ v16bf load_a16x32(const bf16* __restrict__ A, int lda,
                                             int row, int k0, int lane) {
  int half = lane >> 4;
  const v8bf* p0 = reinterpret_cast<const v8bf*>(A + (long)row * lda + k0 + half * 8);
  const v8bf* p1 = reinterpret_cast<const v8bf*>(A + (long)row * lda + k0 + 16 + half * 8);
  v8bf lo = *p0, hi = *p1;
  return __builtin_shufflevector(lo, hi, 0,1,2,3,4,5,6,7,8,9,10,11,12,13,14,15);
}
__device__ __forceinline__ v16bf load_b32x16(const bf16* __restrict__ W, int ldw,
                                             int n, int k0, int lane) {
  int half = lane >> 4;
  const v8bf* p0 = reinterpret_cast<const v8bf*>(W + (long)n * ldw + k0 + half * 8);
  const v8bf* p1 = reinterpret_cast<const v8bf*>(W + (long)n * ldw + k0 + 16 + half * 8);
  v8bf lo = *p0, hi = *p1;
  return __builtin_shufflevector(lo, hi, 0,1,2,3,4,5,6,7,8,9,10,11,12,13,14,15);
}

// ---------------------------------------------------------------------------
// Generic GEMM:  C[M,N] = act( A(MxK,bf16) @ W(NxK,bf16)^T + bias[N] ), f32 out
// Requires lda == K (true at every call site), K in {128, 1024}, K % 32 == 0.
// The 16xK A tile is shared by all 8 waves of the block -> stage it once into
// LDS with CDNA5 async global->LDS transfers (ASYNCcnt), then feed WMMA A
// fragments from LDS while B streams from global/L2.
// grid: x = N/128 (8 waves * 16 cols), y = M/16 ; block = 256 (8 waves)
// act: 0 = none, 1 = tanh
// ---------------------------------------------------------------------------
__global__ __launch_bounds__(256)
void gemm_bias_act(const bf16* __restrict__ A, int lda,
                   const bf16* __restrict__ W, int ldw,
                   const float* __restrict__ bias,
                   float* __restrict__ C, int ldc,
                   int M, int N, int K, int act)
{
  __shared__ __align__(16) bf16 As[16 * 1024];   // up to 32 KB (K<=1024)

  int m0 = blockIdx.y * 16;

  // ---- async stage of the contiguous 16xK A tile (lda == K) ----
  {
    const bf16* src = A + (size_t)m0 * lda;
    int n128 = (16 * K) >> 3;                    // # of 16-byte chunks
    unsigned lbase = (unsigned)(uintptr_t)(&As[0]);
    unsigned long long gbase = (unsigned long long)(uintptr_t)src;
    for (int i = threadIdx.x; i < n128; i += 256) {
      unsigned lds = lbase + ((unsigned)i << 4);
      unsigned long long ga = gbase + ((unsigned long long)i << 4);
      asm volatile("global_load_async_to_lds_b128 %0, %1, off"
                   :: "v"(lds), "v"(ga) : "memory");
    }
    asm volatile("s_wait_asynccnt 0" ::: "memory");
  }
  __syncthreads();

  int lane = threadIdx.x & 31;
  int wave = threadIdx.x >> 5;
  int n0 = blockIdx.x * 128 + wave * 16;
  if (n0 >= N || m0 >= M) return;       // wave-uniform
  int ln = lane & 15, half = lane >> 4;
  int wrow = n0 + ln;                   // output column / W row
  v8f acc = {};
  for (int k0 = 0; k0 < K; k0 += 32) {
    // A fragment from LDS (row = ln, packed row stride K)
    const v8bf* p0 = reinterpret_cast<const v8bf*>(&As[ln * K + k0 + half * 8]);
    const v8bf* p1 = reinterpret_cast<const v8bf*>(&As[ln * K + k0 + 16 + half * 8]);
    v16bf a = __builtin_shufflevector(*p0, *p1, 0,1,2,3,4,5,6,7,8,9,10,11,12,13,14,15);
    v16bf b = load_b32x16(W, ldw, wrow, k0, lane);
    acc = __builtin_amdgcn_wmma_f32_16x16x32_bf16(false, a, false, b,
                                                  (short)0, acc, false, false);
  }
  float bv = bias ? bias[wrow] : 0.0f;
  #pragma unroll
  for (int v = 0; v < 8; ++v) {
    float x = acc[v] + bv;
    if (act == 1) x = tanhf(x);
    C[(long)(m0 + v + 8 * half) * ldc + wrow] = x;
  }
}

// ---------------------------------------------------------------------------
// Fused recurrent GRU step: gh = h @ Whh^T + bhh (WMMA -> LDS), then gates,
// masked state update, output write.  grid = 8 blocks (dir = bx>>2, 16-row
// M tile = (bx&3)*16), block = 256.  LDS: 16 x 1536 f32 = 96 KB.
// ---------------------------------------------------------------------------
__global__ __launch_bounds__(256)
void gru_step(float* __restrict__ h_f32, bf16* __restrict__ h_bf16,
              const bf16* __restrict__ whh2,           // [2][G_*H_] f then b
              const float* __restrict__ bhh_f, const float* __restrict__ bhh_b,
              const float* __restrict__ gi2,           // [2][B_*L_*G_]
              float* __restrict__ out2,                // [2][B_*L_*H_]
              const int* __restrict__ lengths, int t)
{
  __shared__ float ghs[16 * G_];
  int dir = blockIdx.x >> 2;
  int mt  = (blockIdx.x & 3) * 16;
  float* h        = h_f32  + (long)dir * (B_ * H_);
  bf16*  hb       = h_bf16 + (long)dir * (B_ * H_);
  const bf16*  Wh = whh2   + (long)dir * (G_ * H_);
  const float* bh = dir ? bhh_b : bhh_f;
  const float* gi = gi2    + (long)dir * ((long)B_ * L_ * G_);
  float* outd     = out2   + (long)dir * ((long)B_ * L_ * H_);

  int lane = threadIdx.x & 31;
  int wave = threadIdx.x >> 5;
  int ln = lane & 15, half = lane >> 4;
  int arow = mt + ln;

  // Preload the 16x512 bf16 h-tile (A operand) once; reuse over 12 N tiles.
  v16bf areg[16];
  #pragma unroll
  for (int kk = 0; kk < 16; ++kk)
    areg[kk] = load_a16x32(hb, H_, arow, kk * 32, lane);

  for (int nt = 0; nt < 12; ++nt) {
    int n0 = (wave * 12 + nt) * 16;
    int wrow = n0 + ln;
    v8f acc = {};
    #pragma unroll
    for (int kk = 0; kk < 16; ++kk) {
      v16bf b = load_b32x16(Wh, H_, wrow, kk * 32, lane);
      acc = __builtin_amdgcn_wmma_f32_16x16x32_bf16(false, areg[kk], false, b,
                                                    (short)0, acc, false, false);
    }
    float bv = bh[wrow];
    #pragma unroll
    for (int v = 0; v < 8; ++v)
      ghs[(v + 8 * half) * G_ + wrow] = acc[v] + bv;
  }
  __syncthreads();

  for (int p = threadIdx.x; p < 16 * H_; p += 256) {
    int mrow = p >> 9;
    int j = p & (H_ - 1);
    int b = mt + mrow;
    int len = lengths[b]; if (len < 1) len = 1;
    float mval = (t < len) ? 1.0f : 0.0f;
    const float* gir = gi + ((long)b * L_ + t) * G_;
    float gr = ghs[mrow * G_ + j];
    float gz = ghs[mrow * G_ + H_ + j];
    float gn = ghs[mrow * G_ + 2 * H_ + j];
    float r = 1.0f / (1.0f + __expf(-(gir[j] + gr)));
    float z = 1.0f / (1.0f + __expf(-(gir[H_ + j] + gz)));
    float n = tanhf(gir[2 * H_ + j] + r * gn);
    float hold = h[(long)b * H_ + j];
    float hnew = (1.0f - z) * n + z * hold;
    float hsel = mval * hnew + (1.0f - mval) * hold;
    h[(long)b * H_ + j]  = hsel;
    hb[(long)b * H_ + j] = (bf16)hsel;
    outd[((long)b * L_ + t) * H_ + j] = hsel * mval;
  }
}

// ---------------------------------------------------------------------------
// Embedding gather (+ per-row reversed copy for the backward direction)
// ---------------------------------------------------------------------------
__global__ void embed_gather(const float* __restrict__ emb,
                             const int* __restrict__ iv,      // (B,L)
                             const int* __restrict__ lengths, // (B,)
                             bf16* __restrict__ xf, bf16* __restrict__ xb)
{
  int p = blockIdx.x * 256 + threadIdx.x;
  if (p >= B_ * L_ * E_) return;
  int e = p & (E_ - 1);
  int t = (p >> 7) & (L_ - 1);
  int b = p >> 13;
  int len = lengths[b]; if (len < 1) len = 1;
  int rt = (t < len) ? (len - 1 - t) : t;
  xf[p] = (bf16)emb[(long)iv[b * L_ + t]  * E_ + e];
  xb[p] = (bf16)emb[(long)iv[b * L_ + rt] * E_ + e];
}

// x1 = concat(of, reverse(obr)); xr1 = reverse(x1) = concat(reverse(of), obr)
__global__ void build_l1_inputs(const float* __restrict__ of, const float* __restrict__ obr,
                                const int* __restrict__ lengths,
                                bf16* __restrict__ xf, bf16* __restrict__ xb)
{
  long p = (long)blockIdx.x * 256 + threadIdx.x;   // B*L*1024
  if (p >= (long)B_ * L_ * 1024) return;
  int hh = (int)(p & 1023);
  int t = (int)((p >> 10) & (L_ - 1));
  int b = (int)(p >> 16);
  int len = lengths[b]; if (len < 1) len = 1;
  int rt = (t < len) ? (len - 1 - t) : t;
  float vf, vb;
  if (hh < H_) { vf = of[((long)b * L_ + t) * H_ + hh];  vb = of[((long)b * L_ + rt) * H_ + hh]; }
  else { int j = hh - H_; vf = obr[((long)b * L_ + rt) * H_ + j]; vb = obr[((long)b * L_ + t) * H_ + j]; }
  xf[p] = (bf16)vf;
  xb[p] = (bf16)vb;
}

// word_output = of1 + reverse(obr1); write to d_out word region + nodeval leaves
__global__ void word_out_kernel(const float* __restrict__ of, const float* __restrict__ obr,
                                const int* __restrict__ lengths,
                                float* __restrict__ wo,        // (B,L,H) for span s
                                float* __restrict__ nodeval)   // (L,B,H)
{
  long p = (long)blockIdx.x * 256 + threadIdx.x;   // B*L*H
  if (p >= (long)B_ * L_ * H_) return;
  int hh = (int)(p & (H_ - 1));
  int t = (int)((p >> 9) & (L_ - 1));
  int b = (int)(p >> 15);
  int len = lengths[b]; if (len < 1) len = 1;
  int rt = (t < len) ? (len - 1 - t) : t;
  float v = of[((long)b * L_ + t) * H_ + hh] + obr[((long)b * L_ + rt) * H_ + hh];
  wo[((long)b * L_ + t) * H_ + hh] = v;
  nodeval[((long)t * B_ + b) * H_ + hh] = v;
}

// ---------------------------------------------------------------------------
// dep_encode internal node: softmax over <=2 children dots, build y=[mix|wv]
// grid = B blocks, 256 threads
// ---------------------------------------------------------------------------
__global__ void dep_mix(const float* __restrict__ wo,       // (B,L,H) span
                        const float* __restrict__ nodeval,  // (L,B,H)
                        int word, int c0, int c1,
                        bf16* __restrict__ y)               // (B,1024)
{
  __shared__ float red0[256], red1[256];
  __shared__ float a0s, a1s;
  int b = blockIdx.x, tid = threadIdx.x;
  const float* wv = wo + ((long)b * L_ + word) * H_;
  const float* v0 = nodeval + ((long)c0 * B_ + b) * H_;
  const float* v1 = (c1 >= 0) ? (nodeval + ((long)c1 * B_ + b) * H_) : v0;
  float d0 = 0.f, d1 = 0.f;
  for (int h = tid; h < H_; h += 256) { float w = wv[h]; d0 += w * v0[h]; d1 += w * v1[h]; }
  red0[tid] = d0; red1[tid] = d1;
  __syncthreads();
  for (int s = 128; s > 0; s >>= 1) {
    if (tid < s) { red0[tid] += red0[tid + s]; red1[tid] += red1[tid + s]; }
    __syncthreads();
  }
  if (tid == 0) {
    if (c1 < 0) { a0s = 1.0f; a1s = 0.0f; }
    else {
      float s0 = red0[0], s1 = red1[0];
      float mx = fmaxf(s0, s1);
      float e0 = __expf(s0 - mx), e1 = __expf(s1 - mx);
      float inv = 1.0f / (e0 + e1);
      a0s = e0 * inv; a1s = e1 * inv;
    }
  }
  __syncthreads();
  float a0 = a0s, a1 = a1s;
  for (int h = tid; h < H_; h += 256) {
    float mix = a0 * v0[h] + ((c1 >= 0) ? a1 * v1[h] : 0.0f);
    y[(long)b * 1024 + h]      = (bf16)mix;
    y[(long)b * 1024 + H_ + h] = (bf16)wv[h];
  }
}

__global__ void copy_root(const float* __restrict__ nvroot, float* __restrict__ spanin, int s)
{
  int p = blockIdx.x * 256 + threadIdx.x;   // B*H
  if (p >= B_ * H_) return;
  int b = p >> 9, h = p & (H_ - 1);
  spanin[((long)b * S_ + s) * H_ + h] = nvroot[(long)b * H_ + h];
}

// ---------------------------------------------------------------------------
// Span self-attention (S=6): PE add, masked scores, softmax, y=[mix|q]
// grid = B blocks, 256 threads
// ---------------------------------------------------------------------------
__global__ void span_attn(const float* __restrict__ spanin, const int* __restrict__ slen,
                          bf16* __restrict__ y)   // (B*S, 1024)
{
  __shared__ float xsp[S_][H_];
  __shared__ float att[S_][S_];
  int b = blockIdx.x, tid = threadIdx.x;
  const float lnd = __logf(10000.0f) / (float)H_;
  for (int p = tid; p < S_ * H_; p += 256) {
    int q = p / H_, h = p % H_;
    float div = __expf(-(float)(2 * (h >> 1)) * lnd);
    float arg = (float)q * div;
    float pe = (h & 1) ? __cosf(arg) : __sinf(arg);
    xsp[q][h] = spanin[((long)b * S_ + q) * H_ + h] + pe;
  }
  __syncthreads();
  int sl = slen[b];
  if (tid < S_ * S_) {
    int q = tid / S_, k = tid % S_;
    float sc = 0.0f;
    for (int h = 0; h < H_; ++h) sc += xsp[q][h] * xsp[k][h];
    if (k >= sl) sc = -1e9f;
    att[q][k] = sc;
  }
  __syncthreads();
  if (tid < S_) {
    float mx = -1e30f;
    for (int k = 0; k < S_; ++k) mx = fmaxf(mx, att[tid][k]);
    float sum = 0.0f;
    for (int k = 0; k < S_; ++k) { float e = __expf(att[tid][k] - mx); att[tid][k] = e; sum += e; }
    float inv = 1.0f / sum;
    for (int k = 0; k < S_; ++k) att[tid][k] *= inv;
  }
  __syncthreads();
  for (int p = tid; p < S_ * H_; p += 256) {
    int q = p / H_, h = p % H_;
    float mix = 0.0f;
    for (int k = 0; k < S_; ++k) mix += att[q][k] * xsp[k][h];
    y[((long)(b * S_ + q)) * 1024 + h]      = (bf16)mix;
    y[((long)(b * S_ + q)) * 1024 + H_ + h] = (bf16)xsp[q][h];
  }
}

__global__ void span_finalize(const float* __restrict__ raw, const int* __restrict__ slen,
                              float* __restrict__ spanout, float* __restrict__ hidden)
{
  int p = blockIdx.x * 256 + threadIdx.x;   // B*S*H
  if (p >= B_ * S_ * H_) return;
  int h = p & (H_ - 1);
  int q = (p >> 9) % S_;
  int b = p / (S_ * H_);
  int sl = slen[b];
  float v = raw[p] * ((q < sl) ? 1.0f : 0.0f);
  spanout[p] = v;
  if (q == sl - 1) hidden[(long)b * H_ + h] = v;
}

// ---------------------------------------------------------------------------
// Utility kernels
// ---------------------------------------------------------------------------
__global__ void f2bf(const float* __restrict__ src, bf16* __restrict__ dst, long n)
{
  long p = (long)blockIdx.x * 256 + threadIdx.x;
  if (p < n) dst[p] = (bf16)src[p];
}
__global__ void init_h(float* __restrict__ hf, bf16* __restrict__ hb)
{
  int p = blockIdx.x * 256 + threadIdx.x;
  if (p < 4 * B_ * H_) { hf[p] = 0.0f; hb[p] = (bf16)0.0f; }
}

// ---------------------------------------------------------------------------
// Host side
// ---------------------------------------------------------------------------
struct DepNode { int word, c0, c1; };

// Mirrors reference build_tree + dep_encode post-order (children before parent)
static int build_post(int lo, int hi, DepNode* out, int* n)
{
  if (lo > hi) return -1;
  int mid = (lo + hi) / 2;
  int l = build_post(lo, mid - 1, out, n);
  int r = build_post(mid + 1, hi, out, n);
  if (l >= 0 || r >= 0) {
    DepNode nd;
    nd.word = mid;
    nd.c0 = (l >= 0) ? l : r;
    nd.c1 = (l >= 0 && r >= 0) ? r : -1;
    out[(*n)++] = nd;
  }
  return mid;
}

extern "C" void kernel_launch(void* const* d_in, const int* in_sizes, int n_in,
                              void* d_out, int out_size, void* d_ws, size_t ws_size,
                              hipStream_t stream)
{
  // ---- locate inputs by size scan (robust to how `tree` was flattened) ----
  int idx_iv = -1, idx_len = -1, idx_sl = -1, idx_emb = -1;
  for (int i = 0; i < n_in; ++i) {
    if (idx_iv < 0 && in_sizes[i] == S_ * B_ * L_) { idx_iv = i; continue; }
    if (idx_iv >= 0 && idx_len < 0 && in_sizes[i] == S_ * B_) { idx_len = i; continue; }
    if (idx_len >= 0 && idx_sl < 0 && in_sizes[i] == B_) { idx_sl = i; continue; }
    if (idx_emb < 0 && in_sizes[i] == V_ * E_) { idx_emb = i; break; }
  }
  if (idx_iv < 0 || idx_len < 0 || idx_sl < 0 || idx_emb < 0) return;

  const int* iv   = (const int*)d_in[idx_iv];
  const int* lens = (const int*)d_in[idx_len];
  const int* sl   = (const int*)d_in[idx_sl];
  const float* emb      = (const float*)d_in[idx_emb + 0];
  const float* span_W   = (const float*)d_in[idx_emb + 1];
  const float* span_b   = (const float*)d_in[idx_emb + 2];
  const float* parent_W = (const float*)d_in[idx_emb + 3];
  const float* parent_b = (const float*)d_in[idx_emb + 4];
  const float *wih[4], *whh[4], *bih[4], *bhh[4];   // order: l0f,l0b,l1f,l1b
  for (int g = 0; g < 4; ++g) {
    wih[g] = (const float*)d_in[idx_emb + 5 + g * 4 + 0];
    whh[g] = (const float*)d_in[idx_emb + 5 + g * 4 + 1];
    bih[g] = (const float*)d_in[idx_emb + 5 + g * 4 + 2];
    bhh[g] = (const float*)d_in[idx_emb + 5 + g * 4 + 3];
  }

  // ---- workspace layout ----
  char* wsp = (char*)d_ws;
  size_t off = 0;
  auto alloc = [&](size_t bytes) -> void* {
    void* p = wsp + off;
    off = (off + bytes + 255) & ~(size_t)255;
    return p;
  };
  const size_t wihN[4] = { (size_t)G_ * E_, (size_t)G_ * E_, (size_t)G_ * 1024, (size_t)G_ * 1024 };
  bf16* wih_bf[4];
  for (int g = 0; g < 4; ++g) wih_bf[g] = (bf16*)alloc(wihN[g] * sizeof(bf16));
  bf16* whh_bf = (bf16*)alloc((size_t)4 * G_ * H_ * sizeof(bf16));
  bf16* pW_bf  = (bf16*)alloc((size_t)H_ * 1024 * sizeof(bf16));
  bf16* sW_bf  = (bf16*)alloc((size_t)H_ * 1024 * sizeof(bf16));
  bf16* xf     = (bf16*)alloc((size_t)B_ * L_ * 1024 * sizeof(bf16));
  bf16* xb     = (bf16*)alloc((size_t)B_ * L_ * 1024 * sizeof(bf16));
  float* gi    = (float*)alloc((size_t)2 * B_ * L_ * G_ * sizeof(float));
  float* outAB = (float*)alloc((size_t)2 * B_ * L_ * H_ * sizeof(float));
  float* hf    = (float*)alloc((size_t)4 * B_ * H_ * sizeof(float));
  bf16*  hb    = (bf16*)alloc((size_t)4 * B_ * H_ * sizeof(bf16));
  float* nodeval = (float*)alloc((size_t)L_ * B_ * H_ * sizeof(float));
  bf16*  ybuf  = (bf16*)alloc((size_t)B_ * S_ * 1024 * sizeof(bf16));
  float* spanin = (float*)alloc((size_t)B_ * S_ * H_ * sizeof(float));
  float* raw    = (float*)alloc((size_t)B_ * S_ * H_ * sizeof(float));
  (void)ws_size;

  // ---- output regions: span_output | word_outputs | span_hidden ----
  float* outp = (float*)d_out;
  float* out_span = outp;
  float* out_word = outp + (size_t)B_ * S_ * H_;
  float* out_hidden = out_word + (size_t)S_ * B_ * L_ * H_;
  (void)out_size;

  // ---- one-time-per-launch weight conversion + state init ----
  for (int g = 0; g < 4; ++g) {
    f2bf<<<(int)((wihN[g] + 255) / 256), 256, 0, stream>>>(wih[g], wih_bf[g], (long)wihN[g]);
    f2bf<<<(int)(((size_t)G_ * H_ + 255) / 256), 256, 0, stream>>>(whh[g], whh_bf + (size_t)g * G_ * H_, (long)G_ * H_);
  }
  f2bf<<<(int)(((size_t)H_ * 1024 + 255) / 256), 256, 0, stream>>>(parent_W, pW_bf, (long)H_ * 1024);
  f2bf<<<(int)(((size_t)H_ * 1024 + 255) / 256), 256, 0, stream>>>(span_W, sW_bf, (long)H_ * 1024);
  init_h<<<(4 * B_ * H_) / 256, 256, 0, stream>>>(hf, hb);

  // ---- deterministic dependency tree (post-order internal nodes) ----
  DepNode post[L_];
  int npost = 0;
  int root = build_post(0, L_ - 1, post, &npost);

  const size_t giStride = (size_t)B_ * L_ * G_;
  const int M = B_ * L_;   // 4096

  for (int s = 0; s < S_; ++s) {
    const int* iv_s = iv + (size_t)s * B_ * L_;
    const int* len_s = lens + (size_t)s * B_;
    float* wo_s = out_word + (size_t)s * B_ * L_ * H_;

    embed_gather<<<(B_ * L_ * E_) / 256, 256, 0, stream>>>(emb, iv_s, len_s, xf, xb);

    // ---- layer 0 ----
    gemm_bias_act<<<dim3(G_ / 128, M / 16), 256, 0, stream>>>(
        xf, E_, wih_bf[0], E_, bih[0], gi, G_, M, G_, E_, 0);
    gemm_bias_act<<<dim3(G_ / 128, M / 16), 256, 0, stream>>>(
        xb, E_, wih_bf[1], E_, bih[1], gi + giStride, G_, M, G_, E_, 0);
    for (int t = 0; t < L_; ++t)
      gru_step<<<8, 256, 0, stream>>>(hf, hb, whh_bf, bhh[0], bhh[1],
                                      gi, outAB, len_s, t);

    build_l1_inputs<<<(int)(((size_t)B_ * L_ * 1024) / 256), 256, 0, stream>>>(
        outAB, outAB + (size_t)B_ * L_ * H_, len_s, xf, xb);

    // ---- layer 1 ----
    gemm_bias_act<<<dim3(G_ / 128, M / 16), 256, 0, stream>>>(
        xf, 1024, wih_bf[2], 1024, bih[2], gi, G_, M, G_, 1024, 0);
    gemm_bias_act<<<dim3(G_ / 128, M / 16), 256, 0, stream>>>(
        xb, 1024, wih_bf[3], 1024, bih[3], gi + giStride, G_, M, G_, 1024, 0);
    for (int t = 0; t < L_; ++t)
      gru_step<<<8, 256, 0, stream>>>(hf + (size_t)2 * B_ * H_, hb + (size_t)2 * B_ * H_,
                                      whh_bf + (size_t)2 * G_ * H_, bhh[2], bhh[3],
                                      gi, outAB, len_s, t);

    word_out_kernel<<<(int)(((size_t)B_ * L_ * H_) / 256), 256, 0, stream>>>(
        outAB, outAB + (size_t)B_ * L_ * H_, len_s, wo_s, nodeval);

    // ---- dep_encode over tree (post-order) ----
    for (int i = 0; i < npost; ++i) {
      dep_mix<<<B_, 256, 0, stream>>>(wo_s, nodeval, post[i].word, post[i].c0, post[i].c1, ybuf);
      gemm_bias_act<<<dim3(H_ / 128, B_ / 16), 256, 0, stream>>>(
          ybuf, 1024, pW_bf, 1024, parent_b,
          nodeval + (size_t)post[i].word * B_ * H_, H_, B_, H_, 1024, 1);
    }
    copy_root<<<(B_ * H_) / 256, 256, 0, stream>>>(nodeval + (size_t)root * B_ * H_, spanin, s);
  }

  // ---- span self-attention ----
  span_attn<<<B_, 256, 0, stream>>>(spanin, sl, ybuf);
  gemm_bias_act<<<dim3(H_ / 128, (B_ * S_) / 16), 256, 0, stream>>>(
      ybuf, 1024, sW_bf, 1024, span_b, raw, H_, B_ * S_, H_, 1024, 1);
  span_finalize<<<(B_ * S_ * H_) / 256, 256, 0, stream>>>(raw, sl, out_span, out_hidden);
}
